// LatentTransition_40492951666850
// MI455X (gfx1250) — matrix operand order repaired and checked by reading, not compiled, not run on hardware
//
#include <hip/hip_runtime.h>
#include <stdint.h>

// ---------------------------------------------------------------------------
// LatentTransition for MI455X (gfx1250, wave32, WMMA).
//
// Roofline: ~420 GFLOP fp32-equiv, ~3-4 GB HBM traffic -> memory bound at
// 23.3 TB/s (~200us floor). Strategy: scatter-mean with f32 global atomics,
// all GEMMs fused per 16-row tile on v_wmma_f32_16x16x32_bf16 using a
// 3-term bf16 hi/lo split (~fp32 accuracy, compute still free vs memory).
// The hi/lo split is applied ONCE per element at LDS-staging time (two
// ushort planes); per-wave fragment builds are pure ds_load_b32 in the ISA
// A-matrix lane layout. Weights are pre-packed into the ISA B-matrix lane
// layout (per-lane-contiguous -> global_load_b128).
// ---------------------------------------------------------------------------

#define LATENT  128
#define HIDDEN  256
#define CE_DIM  32
#define CTRL_D  16
#define NLAYERS 2
#define N1D     20000
#define N2D     100000
#define BATCH   4

typedef __attribute__((ext_vector_type(16))) __bf16          v16bf;
typedef __attribute__((ext_vector_type(8)))  float           v8f;
typedef __attribute__((ext_vector_type(8)))  unsigned int    v8u;

__device__ __forceinline__ unsigned short f32_to_bf16_rne(float f) {
  unsigned u = __builtin_bit_cast(unsigned, f);
  u += 0x7fffu + ((u >> 16) & 1u);
  return (unsigned short)(u >> 16);
}
__device__ __forceinline__ float bf16_to_f32(unsigned short h) {
  unsigned u = ((unsigned)h) << 16;
  return __builtin_bit_cast(float, u);
}
// split x ~= hi + lo (two bf16 terms, ~16 mantissa bits total)
__device__ __forceinline__ void cvt_split(float x, unsigned short& h, unsigned short& l) {
  h = f32_to_bf16_rne(x);
  l = f32_to_bf16_rne(x - bf16_to_f32(h));
}

// D = A*B + C  with 3-term bf16 split (Ah*Bh + Ah*Bl + Al*Bh)
__device__ __forceinline__ v8f wmma3(v16bf ah, v16bf al, v16bf bh, v16bf bl, v8f acc) {
  acc = __builtin_amdgcn_wmma_f32_16x16x32_bf16(false, ah, false, bh, (short)0, acc, false, false);
  acc = __builtin_amdgcn_wmma_f32_16x16x32_bf16(false, ah, false, bl, (short)0, acc, false, false);
  acc = __builtin_amdgcn_wmma_f32_16x16x32_bf16(false, al, false, bh, (short)0, acc, false, false);
  return acc;
}

// Build A fragment (16x32) from pre-split bf16 planes in LDS.
// ISA 16-bit A layout: row = lane&15, half = lane>>4;
//   vgpr j<4 : K = half*8 + 2j,2j+1 ; vgpr j>=4 : K = 16 + half*8 + 2(j-4),+1
// k0 is even, so one aligned dword load delivers the packed (K,K+1) pair.
__device__ __forceinline__ void load_afrag_lds(const unsigned short* Ah, const unsigned short* Al,
                                               int stride, int lane, v16bf& ah, v16bf& al) {
  const int row = lane & 15, half = lane >> 4;
  v8u hv, lv;
#pragma unroll
  for (int j = 0; j < 8; ++j) {
    const int k0 = (j < 4) ? (half * 8 + 2 * j) : (16 + half * 8 + 2 * (j - 4));
    hv[j] = *(const unsigned*)(Ah + row * stride + k0);
    lv[j] = *(const unsigned*)(Al + row * stride + k0);
  }
  ah = __builtin_bit_cast(v16bf, hv);
  al = __builtin_bit_cast(v16bf, lv);
}

// Load pre-packed B fragment: layout [((nt*KS+ks)*32+lane)*16 + v], v0..7=hi
// pairs, v8..15=lo pairs -> two 32B per-lane loads.
__device__ __forceinline__ void load_wfrag(const unsigned int* __restrict__ wpk,
                                           int nt, int ks, int KS, int lane,
                                           v16bf& bh, v16bf& bl) {
  const v8u* p = (const v8u*)(wpk + (((size_t)nt * KS + ks) * 32 + lane) * 16);
  v8u hw = p[0];
  v8u lw = p[1];
  bh = __builtin_bit_cast(v16bf, hw);
  bl = __builtin_bit_cast(v16bf, lw);
}

// C/D layout: n = lane&15, vgpr i -> m = i + 8*(lane>>4)
__device__ __forceinline__ void store_acc_f32(float* dst, int stride, int nbase, int lane, v8f acc) {
  const int n = lane & 15, half = lane >> 4;
#pragma unroll
  for (int i = 0; i < 8; ++i) dst[(i + 8 * half) * stride + nbase + n] = acc[i];
}

// store accumulator tile split into bf16 hi/lo LDS planes (consumed as A)
__device__ __forceinline__ void store_acc_split(unsigned short* dh, unsigned short* dl,
                                                int stride, int nbase, int lane, v8f acc) {
  const int n = lane & 15, half = lane >> 4;
#pragma unroll
  for (int i = 0; i < 8; ++i) {
    unsigned short h, l;
    cvt_split(acc[i], h, l);
    dh[(i + 8 * half) * stride + nbase + n] = h;
    dl[(i + 8 * half) * stride + nbase + n] = l;
  }
}

// ------------------------------ small kernels ------------------------------

__global__ void zero_kernel(float* __restrict__ p, size_t n) {
  size_t i = (size_t)blockIdx.x * blockDim.x + threadIdx.x;
  if (i < n) p[i] = 0.f;
}

__global__ void copy_kernel(float* __restrict__ dst, const float* __restrict__ src, size_t n) {
  size_t i = (size_t)blockIdx.x * blockDim.x + threadIdx.x;
  if (i < n) dst[i] = src[i];
}

__global__ void bias_add_kernel(float* __restrict__ out, const float* __restrict__ a,
                                const float* __restrict__ b, int n) {
  int i = blockIdx.x * blockDim.x + threadIdx.x;
  if (i < n) out[i] = a[i] + b[i];
}

__global__ void count_kernel(const int* __restrict__ dst, int e, float* __restrict__ cnt) {
  int i = blockIdx.x * blockDim.x + threadIdx.x;
  if (i < e) atomicAdd(&cnt[dst[i]], 1.0f);
}

// scatter-add messages: acat base already includes the 128-col slot offset;
// row stride of acat is 256 floats.
__global__ void scatter_kernel(const float* __restrict__ src, float* __restrict__ acat,
                               const int* __restrict__ ei0, const int* __restrict__ ei1,
                               int e, int nsrc, int ndst) {
  size_t tid = (size_t)blockIdx.x * blockDim.x + threadIdx.x;
  size_t tot = (size_t)BATCH * e * 32;
  if (tid >= tot) return;
  const int chunk = (int)(tid & 31);
  const size_t t2 = tid >> 5;
  const int ed = (int)(t2 % (size_t)e);
  const int b  = (int)(t2 / (size_t)e);
  const size_t srow = (size_t)b * nsrc + ei0[ed];
  const size_t drow = (size_t)b * ndst + ei1[ed];
  const float4 v = ((const float4*)(src + srow * 128))[chunk];
  float* d = acat + drow * 256 + chunk * 4;
  atomicAdd(d + 0, v.x);
  atomicAdd(d + 1, v.y);
  atomicAdd(d + 2, v.z);
  atomicAdd(d + 3, v.w);
}

__global__ void finalize_mean_kernel(float* __restrict__ acat, const float* __restrict__ cntA,
                                     const float* __restrict__ cntB, size_t m, int nnode) {
  size_t tid = (size_t)blockIdx.x * blockDim.x + threadIdx.x;
  size_t tot = m * 256;
  if (tid >= tot) return;
  const size_t row = tid >> 8;
  const int c = (int)(tid & 255);
  const int node = (int)(row % (size_t)nnode);
  const float cnt = (c < 128) ? cntA[node] : cntB[node];
  acat[tid] = acat[tid] / fmaxf(cnt, 1.f);
}

// ----------------------------- weight packing ------------------------------
// Pack W[N][K] (row-major, out = A @ W^T) into the ISA bf16 B-matrix layout:
// j<4: K = half*8+2j,+1 ; j>=4: K = 16+half*8+2(j-4),+1 ; col = nt*16 + (lane&15).
// Output dword index = ((nt*KS+ks)*32+lane)*16 + v, v<8 hi pair j=v, v>=8 lo.

__global__ void pack_cat3_kernel(unsigned int* __restrict__ out,
                                 const float* __restrict__ wla, const float* __restrict__ wlb,
                                 const float* __restrict__ wra, const float* __restrict__ wrb) {
  const int tid = blockIdx.x * blockDim.x + threadIdx.x;
  const int TOT = 16 * 12 * 32 * 16;           // N=256 (16 ntiles), K=384 (12 ksteps)
  if (tid >= TOT) return;
  const int v    = tid & 15;
  const int lane = (tid >> 4) & 31;
  const int rest = tid >> 9;
  const int ks = rest % 12, nt = rest / 12;
  const int j = v & 7;
  const bool lo = v >= 8;
  const int k0  = (j < 4) ? ((lane >> 4) * 8 + 2 * j) : (16 + (lane >> 4) * 8 + 2 * (j - 4));
  const int col = nt * 16 + (lane & 15);
  const int kk  = ks * 32 + k0;
  auto fetch = [&](int k) -> float {
    if (k < 128) return wla[col * 128 + k];
    if (k < 256) return wlb[col * 128 + (k - 128)];
    return wra[col * 128 + (k - 256)] + wrb[col * 128 + (k - 256)];
  };
  const float w0 = fetch(kk), w1 = fetch(kk + 1);
  unsigned short h0, l0, h1, l1;
  cvt_split(w0, h0, l0);
  cvt_split(w1, h1, l1);
  out[tid] = lo ? ((unsigned)l0 | ((unsigned)l1 << 16))
                : ((unsigned)h0 | ((unsigned)h1 << 16));
}

__global__ void pack_single_kernel(unsigned int* __restrict__ out, const float* __restrict__ w,
                                   int NT, int KS, int kreal) {
  const int tid = blockIdx.x * blockDim.x + threadIdx.x;
  const int TOT = NT * KS * 512;
  if (tid >= TOT) return;
  const int v    = tid & 15;
  const int lane = (tid >> 4) & 31;
  const int rest = tid >> 9;
  const int ks = rest % KS, nt = rest / KS;
  const int j = v & 7;
  const bool lo = v >= 8;
  const int k0  = (j < 4) ? ((lane >> 4) * 8 + 2 * j) : (16 + (lane >> 4) * 8 + 2 * (j - 4));
  const int col = nt * 16 + (lane & 15);
  const int kk  = ks * 32 + k0;
  const float w0 = (kk     < kreal) ? w[(size_t)col * kreal + kk]     : 0.f;
  const float w1 = (kk + 1 < kreal) ? w[(size_t)col * kreal + kk + 1] : 0.f;
  unsigned short h0, l0, h1, l1;
  cvt_split(w0, h0, l0);
  cvt_split(w1, h1, l1);
  out[tid] = lo ? ((unsigned)l0 | ((unsigned)l1 << 16))
                : ((unsigned)h0 | ((unsigned)h1 << 16));
}

// ----------------------- fused SAGE + proj + LN kernel ---------------------
// One block = 16 rows. Phase1: Hid[16,256] = [mean0|mean1|z]@Wcat^T + bias
// (K=384). Phase2: Dt[16,128] = Hid@proj^T + projb. Phase3: LayerNorm per
// row, dsp += d, z += d.
__global__ __launch_bounds__(256) void conv_gemm_kernel(
    const float* __restrict__ acat, float* __restrict__ z, float* __restrict__ dsp,
    const unsigned int* __restrict__ wpk1, const float* __restrict__ bias1,
    const unsigned int* __restrict__ wpk2, const float* __restrict__ projb,
    const float* __restrict__ lng, const float* __restrict__ lnb) {
  __shared__ unsigned short Ash[16 * 32];
  __shared__ unsigned short Asl[16 * 32];
  __shared__ unsigned short Hidh[16 * HIDDEN];
  __shared__ unsigned short Hidl[16 * HIDDEN];
  __shared__ float Dt[16 * LATENT];
  const int tid = threadIdx.x, lane = tid & 31, wave = tid >> 5;
  const size_t row0 = (size_t)blockIdx.x * 16;

  // phase 1
  v8f acc0, acc1;
  {
    const float b0 = bias1[wave * 32 + (lane & 15)];
    const float b1 = bias1[wave * 32 + 16 + (lane & 15)];
#pragma unroll
    for (int i = 0; i < 8; ++i) { acc0[i] = b0; acc1[i] = b1; }
  }
  for (int ks = 0; ks < 12; ++ks) {
    __syncthreads();
#pragma unroll
    for (int e = 0; e < 2; ++e) {
      const int idx = tid + e * 256;
      const int r = idx >> 5, c = idx & 31;
      float val;
      if (ks < 8) val = acat[(row0 + r) * 256 + (size_t)ks * 32 + c];
      else        val = z[(row0 + r) * 128 + (size_t)(ks - 8) * 32 + c];
      unsigned short h, l;
      cvt_split(val, h, l);
      Ash[idx] = h;
      Asl[idx] = l;
    }
    if (ks + 1 < 8)
      __builtin_prefetch(&acat[(row0 + (tid >> 5)) * 256 + (size_t)(ks + 1) * 32], 0, 1);
    __syncthreads();
    v16bf ah, al, bh, bl;
    load_afrag_lds(Ash, Asl, 32, lane, ah, al);
    load_wfrag(wpk1, wave * 2, ks, 12, lane, bh, bl);
    acc0 = wmma3(ah, al, bh, bl, acc0);
    load_wfrag(wpk1, wave * 2 + 1, ks, 12, lane, bh, bl);
    acc1 = wmma3(ah, al, bh, bl, acc1);
  }
  store_acc_split(Hidh, Hidl, HIDDEN, wave * 32, lane, acc0);
  store_acc_split(Hidh, Hidl, HIDDEN, wave * 32 + 16, lane, acc1);
  __syncthreads();

  // phase 2: proj (K=256, N=128); wave w handles columns [16w,16w+16)
  v8f pacc;
  {
    const float b = projb[wave * 16 + (lane & 15)];
#pragma unroll
    for (int i = 0; i < 8; ++i) pacc[i] = b;
  }
#pragma unroll
  for (int ks = 0; ks < 8; ++ks) {
    v16bf ah, al, bh, bl;
    load_afrag_lds(&Hidh[ks * 32], &Hidl[ks * 32], HIDDEN, lane, ah, al);
    load_wfrag(wpk2, wave, ks, 8, lane, bh, bl);
    pacc = wmma3(ah, al, bh, bl, pacc);
  }
  store_acc_f32(Dt, LATENT, wave * 16, lane, pacc);
  __syncthreads();

  // phase 3: per-row LayerNorm + residual. wave w owns rows 2w, 2w+1.
#pragma unroll
  for (int rr = 0; rr < 2; ++rr) {
    const int r = wave * 2 + rr;
    float x[4];
    float s = 0.f;
#pragma unroll
    for (int q = 0; q < 4; ++q) { x[q] = Dt[r * LATENT + lane + 32 * q]; s += x[q]; }
#pragma unroll
    for (int m = 16; m >= 1; m >>= 1) s += __shfl_xor(s, m, 32);
    const float mu = s * (1.f / 128.f);
    float var = 0.f;
#pragma unroll
    for (int q = 0; q < 4; ++q) { const float d = x[q] - mu; var += d * d; }
#pragma unroll
    for (int m = 16; m >= 1; m >>= 1) var += __shfl_xor(var, m, 32);
    const float rstd = rsqrtf(var * (1.f / 128.f) + 1e-5f);
    const size_t base = (row0 + r) * LATENT;
#pragma unroll
    for (int q = 0; q < 4; ++q) {
      const int c = lane + 32 * q;
      const float d = (x[q] - mu) * rstd * lng[c] + lnb[c];
      dsp[base + c] += d;
      z[base + c] += d;
    }
  }
}

// --------------------- fused MLP + gate + output kernel --------------------
// in = [z0 | ce | u] (K padded 176->192); hid = gelu(in@W1^T+b1);
// del = hid@W2^T+b2 + dsp; g = sigmoid([z0|del]@Wg^T+gb); out = z0 + g*del.
__global__ __launch_bounds__(256) void mlp_gate_kernel(
    const float* __restrict__ z0, const float* __restrict__ ce, const float* __restrict__ u,
    const float* __restrict__ dsp,
    const unsigned int* __restrict__ w1pk, const float* __restrict__ b1,
    const unsigned int* __restrict__ w2pk, const float* __restrict__ b2,
    const unsigned int* __restrict__ gpk, const float* __restrict__ gb,
    float* __restrict__ out, int nnode) {
  __shared__ unsigned short Ash[16 * 32];
  __shared__ unsigned short Asl[16 * 32];
  __shared__ unsigned short Hidh[16 * HIDDEN];
  __shared__ unsigned short Hidl[16 * HIDDEN];
  __shared__ float Zsf[16 * LATENT];
  __shared__ unsigned short Zsh[16 * LATENT];
  __shared__ unsigned short Zsl[16 * LATENT];
  __shared__ float Delf[16 * LATENT];
  __shared__ unsigned short Delh[16 * LATENT];
  __shared__ unsigned short Dell[16 * LATENT];
  const int tid = threadIdx.x, lane = tid & 31, wave = tid >> 5;
  const size_t row0 = (size_t)blockIdx.x * 16;

#pragma unroll
  for (int e = 0; e < 8; ++e) {
    const int idx = tid + e * 256;
    const float zv = z0[row0 * 128 + idx];
    Zsf[idx] = zv;
    unsigned short h, l;
    cvt_split(zv, h, l);
    Zsh[idx] = h;
    Zsl[idx] = l;
  }

  // phase 1: K=192 (6 ksteps), N=256
  v8f acc0, acc1;
  {
    const float c0 = b1[wave * 32 + (lane & 15)];
    const float c1 = b1[wave * 32 + 16 + (lane & 15)];
#pragma unroll
    for (int i = 0; i < 8; ++i) { acc0[i] = c0; acc1[i] = c1; }
  }
  for (int ks = 0; ks < 6; ++ks) {
    __syncthreads();
#pragma unroll
    for (int e = 0; e < 2; ++e) {
      const int idx = tid + e * 256;
      const int r = idx >> 5, c = idx & 31;
      const int k = ks * 32 + c;
      const size_t grow = row0 + r;
      float val;
      if (k < 128)      val = Zsf[r * 128 + k];
      else if (k < 160) val = ce[(grow / (size_t)nnode) * CE_DIM + (k - 128)];
      else if (k < 176) val = u[grow * CTRL_D + (k - 160)];
      else              val = 0.f;
      unsigned short h, l;
      cvt_split(val, h, l);
      Ash[idx] = h;
      Asl[idx] = l;
    }
    __syncthreads();
    v16bf ah, al, bh, bl;
    load_afrag_lds(Ash, Asl, 32, lane, ah, al);
    load_wfrag(w1pk, wave * 2, ks, 6, lane, bh, bl);
    acc0 = wmma3(ah, al, bh, bl, acc0);
    load_wfrag(w1pk, wave * 2 + 1, ks, 6, lane, bh, bl);
    acc1 = wmma3(ah, al, bh, bl, acc1);
  }
  // gelu (tanh approximation, matches jax.nn.gelu default)
#pragma unroll
  for (int i = 0; i < 8; ++i) {
    float x = acc0[i];
    acc0[i] = 0.5f * x * (1.f + tanhf(0.7978845608f * (x + 0.044715f * x * x * x)));
    x = acc1[i];
    acc1[i] = 0.5f * x * (1.f + tanhf(0.7978845608f * (x + 0.044715f * x * x * x)));
  }
  store_acc_split(Hidh, Hidl, HIDDEN, wave * 32, lane, acc0);
  store_acc_split(Hidh, Hidl, HIDDEN, wave * 32 + 16, lane, acc1);
  __syncthreads();

  // phase 2: K=256, N=128 ; del = result + dsp
  v8f pacc;
  {
    const float b = b2[wave * 16 + (lane & 15)];
#pragma unroll
    for (int i = 0; i < 8; ++i) pacc[i] = b;
  }
#pragma unroll
  for (int ks = 0; ks < 8; ++ks) {
    v16bf ah, al, bh, bl;
    load_afrag_lds(&Hidh[ks * 32], &Hidl[ks * 32], HIDDEN, lane, ah, al);
    load_wfrag(w2pk, wave, ks, 8, lane, bh, bl);
    pacc = wmma3(ah, al, bh, bl, pacc);
  }
  {
    const int n = lane & 15, half = lane >> 4;
#pragma unroll
    for (int i = 0; i < 8; ++i) {
      const int m = i + 8 * half;
      const int col = wave * 16 + n;
      const float dv = pacc[i] + dsp[(row0 + m) * LATENT + col];
      Delf[m * LATENT + col] = dv;
      unsigned short h, l;
      cvt_split(dv, h, l);
      Delh[m * LATENT + col] = h;
      Dell[m * LATENT + col] = l;
    }
  }
  __syncthreads();

  // phase 3: gate GEMM, A = [z0 | del], K=256, N=128
  v8f gacc;
  {
    const float b = gb[wave * 16 + (lane & 15)];
#pragma unroll
    for (int i = 0; i < 8; ++i) gacc[i] = b;
  }
#pragma unroll
  for (int ks = 0; ks < 8; ++ks) {
    v16bf ah, al, bh, bl;
    if (ks < 4) load_afrag_lds(&Zsh[ks * 32], &Zsl[ks * 32], LATENT, lane, ah, al);
    else        load_afrag_lds(&Delh[(ks - 4) * 32], &Dell[(ks - 4) * 32], LATENT, lane, ah, al);
    load_wfrag(gpk, wave, ks, 8, lane, bh, bl);
    gacc = wmma3(ah, al, bh, bl, gacc);
  }
  {
    const int n = lane & 15, half = lane >> 4;
#pragma unroll
    for (int i = 0; i < 8; ++i) {
      const int m = i + 8 * half;
      const int col = wave * 16 + n;
      const float g = 1.f / (1.f + expf(-gacc[i]));
      out[(row0 + m) * LATENT + col] = Zsf[m * LATENT + col] + g * Delf[m * LATENT + col];
    }
  }
}

// ------------------------------- host side ---------------------------------

static inline unsigned cdiv(size_t n, int t) { return (unsigned)((n + t - 1) / t); }

extern "C" void kernel_launch(void* const* d_in, const int* in_sizes, int n_in,
                              void* d_out, int out_size, void* d_ws, size_t ws_size,
                              hipStream_t stream) {
  const float* z_1d = (const float*)d_in[0];
  const float* z_2d = (const float*)d_in[1];
  const float* u_1d = (const float*)d_in[2];
  const float* u_2d = (const float*)d_in[3];
  const float* c_e  = (const float*)d_in[4];
  const int* ei_pipe = (const int*)d_in[5];
  const int* ei_surf = (const int*)d_in[6];
  const int* ei_c2   = (const int*)d_in[7];
  const int* ei_c1   = (const int*)d_in[8];
  const float* sage_Wl = (const float*)d_in[9];
  const float* sage_bl = (const float*)d_in[10];
  const float* sage_Wr = (const float*)d_in[11];
  const float* proj_W  = (const float*)d_in[12];
  const float* proj_b  = (const float*)d_in[13];
  const float* ln_g    = (const float*)d_in[14];
  const float* ln_b    = (const float*)d_in[15];
  const float* mlp_W1  = (const float*)d_in[16];
  const float* mlp_b1  = (const float*)d_in[17];
  const float* mlp_W2  = (const float*)d_in[18];
  const float* mlp_b2  = (const float*)d_in[19];
  const float* gate_W  = (const float*)d_in[20];
  const float* gate_b  = (const float*)d_in[21];

  const int Ep  = in_sizes[5] / 2;   // 40000
  const int Es  = in_sizes[6] / 2;   // 400000
  const int Ec2 = in_sizes[7] / 2;   // 20000
  const int Ec1 = in_sizes[8] / 2;   // 20000

  const size_t M1 = (size_t)BATCH * N1D;   // 80000
  const size_t M2 = (size_t)BATCH * N2D;   // 400000

  // ---- workspace carve ----
  float* w = (float*)d_ws;
  auto alloc = [&](size_t n) { float* p = w; w += n; return p; };
  float* z1c  = alloc(M1 * 128);
  float* z2c  = alloc(M2 * 128);
  float* dsp1 = alloc(M1 * 128);
  float* dsp2 = alloc(M2 * 128);
  float* acat1 = alloc(M1 * 256);
  float* acat2 = alloc(M2 * 256);
  float* cntPipe = alloc(N1D);
  float* cntC1   = alloc(N1D);
  float* cntSurf = alloc(N2D);
  float* cntC2   = alloc(N2D);
  unsigned int* wcatPk[NLAYERS][2];
  unsigned int* projPk[NLAYERS][2];
  float* bcat[NLAYERS][2];
  for (int l = 0; l < NLAYERS; ++l)
    for (int t = 0; t < 2; ++t) {
      wcatPk[l][t] = (unsigned int*)alloc(16 * 12 * 512);
      projPk[l][t] = (unsigned int*)alloc(8 * 8 * 512);
      bcat[l][t]   = alloc(256);
    }
  unsigned int* w1Pk[2];
  unsigned int* w2Pk[2];
  unsigned int* gPk[2];
  for (int t = 0; t < 2; ++t) {
    w1Pk[t] = (unsigned int*)alloc(16 * 6 * 512);
    w2Pk[t] = (unsigned int*)alloc(8 * 8 * 512);
    gPk[t]  = (unsigned int*)alloc(8 * 8 * 512);
  }

  // ---- edge counts (layer-invariant) ----
  {
    const size_t n = (size_t)(2 * N1D + 2 * N2D);
    zero_kernel<<<cdiv(n, 256), 256, 0, stream>>>(cntPipe, n);
  }
  count_kernel<<<cdiv((size_t)Ep, 256), 256, 0, stream>>>(ei_pipe + Ep, Ep, cntPipe);
  count_kernel<<<cdiv((size_t)Ec1, 256), 256, 0, stream>>>(ei_c1 + Ec1, Ec1, cntC1);
  count_kernel<<<cdiv((size_t)Es, 256), 256, 0, stream>>>(ei_surf + Es, Es, cntSurf);
  count_kernel<<<cdiv((size_t)Ec2, 256), 256, 0, stream>>>(ei_c2 + Ec2, Ec2, cntC2);

  // ---- init latents + dsp ----
  copy_kernel<<<cdiv(M1 * 128, 256), 256, 0, stream>>>(z1c, z_1d, M1 * 128);
  copy_kernel<<<cdiv(M2 * 128, 256), 256, 0, stream>>>(z2c, z_2d, M2 * 128);
  zero_kernel<<<cdiv((M1 + M2) * 128, 256), 256, 0, stream>>>(dsp1, (M1 + M2) * 128);

  // ---- pack all weights up front (data-independent) ----
  const size_t WSZ = (size_t)HIDDEN * LATENT;        // 256*128
  const size_t PSZ = (size_t)LATENT * HIDDEN;        // 128*256
  for (int l = 0; l < NLAYERS; ++l) {
    // type 0 (1d dst): edge types pipe(0) + c1(3)
    pack_cat3_kernel<<<cdiv(16 * 12 * 512, 256), 256, 0, stream>>>(
        wcatPk[l][0],
        sage_Wl + ((size_t)l * 4 + 0) * WSZ, sage_Wl + ((size_t)l * 4 + 3) * WSZ,
        sage_Wr + ((size_t)l * 4 + 0) * WSZ, sage_Wr + ((size_t)l * 4 + 3) * WSZ);
    bias_add_kernel<<<1, 256, 0, stream>>>(bcat[l][0],
        sage_bl + ((size_t)l * 4 + 0) * 256, sage_bl + ((size_t)l * 4 + 3) * 256, 256);
    // type 1 (2d dst): edge types surf(1) + c2(2)
    pack_cat3_kernel<<<cdiv(16 * 12 * 512, 256), 256, 0, stream>>>(
        wcatPk[l][1],
        sage_Wl + ((size_t)l * 4 + 1) * WSZ, sage_Wl + ((size_t)l * 4 + 2) * WSZ,
        sage_Wr + ((size_t)l * 4 + 1) * WSZ, sage_Wr + ((size_t)l * 4 + 2) * WSZ);
    bias_add_kernel<<<1, 256, 0, stream>>>(bcat[l][1],
        sage_bl + ((size_t)l * 4 + 1) * 256, sage_bl + ((size_t)l * 4 + 2) * 256, 256);
    for (int t = 0; t < 2; ++t)
      pack_single_kernel<<<cdiv(8 * 8 * 512, 256), 256, 0, stream>>>(
          projPk[l][t], proj_W + ((size_t)l * 2 + t) * PSZ, 8, 8, 256);
  }
  for (int t = 0; t < 2; ++t) {
    pack_single_kernel<<<cdiv(16 * 6 * 512, 256), 256, 0, stream>>>(
        w1Pk[t], mlp_W1 + (size_t)t * 256 * 176, 16, 6, 176);
    pack_single_kernel<<<cdiv(8 * 8 * 512, 256), 256, 0, stream>>>(
        w2Pk[t], mlp_W2 + (size_t)t * 128 * 256, 8, 8, 256);
    pack_single_kernel<<<cdiv(8 * 8 * 512, 256), 256, 0, stream>>>(
        gPk[t], gate_W + (size_t)t * 128 * 256, 8, 8, 256);
  }

  // ---- layers ----
  for (int l = 0; l < NLAYERS; ++l) {
    zero_kernel<<<cdiv((M1 + M2) * 256, 256), 256, 0, stream>>>(acat1, (M1 + M2) * 256);
    // means: acat1 slot0 = pipe (src z1c), slot1 = c1 (src z2c)
    scatter_kernel<<<cdiv((size_t)BATCH * Ep * 32, 256), 256, 0, stream>>>(
        z1c, acat1 + 0, ei_pipe, ei_pipe + Ep, Ep, N1D, N1D);
    scatter_kernel<<<cdiv((size_t)BATCH * Ec1 * 32, 256), 256, 0, stream>>>(
        z2c, acat1 + 128, ei_c1, ei_c1 + Ec1, Ec1, N2D, N1D);
    // acat2 slot0 = surf (src z2c), slot1 = c2 (src z1c)
    scatter_kernel<<<cdiv((size_t)BATCH * Es * 32, 256), 256, 0, stream>>>(
        z2c, acat2 + 0, ei_surf, ei_surf + Es, Es, N2D, N2D);
    scatter_kernel<<<cdiv((size_t)BATCH * Ec2 * 32, 256), 256, 0, stream>>>(
        z1c, acat2 + 128, ei_c2, ei_c2 + Ec2, Ec2, N1D, N2D);
    finalize_mean_kernel<<<cdiv(M1 * 256, 256), 256, 0, stream>>>(acat1, cntPipe, cntC1, M1, N1D);
    finalize_mean_kernel<<<cdiv(M2 * 256, 256), 256, 0, stream>>>(acat2, cntSurf, cntC2, M2, N2D);
    // fused GEMM + proj + LN + residual
    conv_gemm_kernel<<<(unsigned)(M1 / 16), 256, 0, stream>>>(
        acat1, z1c, dsp1, wcatPk[l][0], bcat[l][0], projPk[l][0],
        proj_b + ((size_t)l * 2 + 0) * 128, ln_g + ((size_t)l * 2 + 0) * 128,
        ln_b + ((size_t)l * 2 + 0) * 128);
    conv_gemm_kernel<<<(unsigned)(M2 / 16), 256, 0, stream>>>(
        acat2, z2c, dsp2, wcatPk[l][1], bcat[l][1], projPk[l][1],
        proj_b + ((size_t)l * 2 + 1) * 128, ln_g + ((size_t)l * 2 + 1) * 128,
        ln_b + ((size_t)l * 2 + 1) * 128);
  }

  // ---- fused MLP + gate + output ----
  float* out = (float*)d_out;
  mlp_gate_kernel<<<(unsigned)(M1 / 16), 256, 0, stream>>>(
      z_1d, c_e, u_1d, dsp1, w1Pk[0], mlp_b1 + 0, w2Pk[0], mlp_b2 + 0,
      gPk[0], gate_b + 0, out, N1D);
  mlp_gate_kernel<<<(unsigned)(M2 / 16), 256, 0, stream>>>(
      z_2d, c_e, u_2d, dsp2, w1Pk[1], mlp_b1 + 256, w2Pk[1], mlp_b2 + 128,
      gPk[1], gate_b + 128, out + M1 * 128, N2D);
}
